// ES_RNN_EX_49778670960919
// MI455X (gfx1250) — matrix operand order repaired and checked by reading, not compiled, not run on hardware
//
#include <hip/hip_runtime.h>
#include <hip/hip_bf16.h>

// ---------------------------------------------------------------------------
// ES-RNN for MI455X (gfx1250, wave32, WMMA).
// Pipeline: es_scan (serial, 1 wave) -> logs -> {lvp, labels, seq, weights}
//           -> fused LSTM+head using v_wmma_f32_16x16x32_f16.
// ---------------------------------------------------------------------------

typedef __attribute__((ext_vector_type(16))) _Float16 v16h;
typedef __attribute__((ext_vector_type(8)))  float    v8f;
typedef __attribute__((ext_vector_type(4)))  unsigned u32x4;

union AF { v16h v; u32x4 q[2]; };

static constexpr int F    = 8;
static constexpr int T    = 16384;
static constexpr int S    = 24;
static constexpr int H    = 64;
static constexpr int Wn   = 168;
static constexpr int On   = 48;
static constexpr int N    = T - Wn - On + 1;   // 16169
static constexpr int ROWS = 64;                // rows of N per block
static constexpr int NPAD = 16192;             // 253 * 64
static constexpr int KC   = 96;                // combined K: 64 (h) + 8 (x) + 24 pad
static constexpr int GC   = 256;               // gate columns (4*H)

// rcp-based activations: no IEEE divide chains (v_exp + v_rcp only).
// Infinity saturates correctly through rcp (rcp(inf) = 0), so no clamps.
__device__ __forceinline__ float sigf(float x) {
    return __builtin_amdgcn_rcpf(1.0f + __expf(-x));
}
__device__ __forceinline__ float tanh_fast(float x) {
    float e = __expf(2.0f * x);                 // tanh = 1 - 2/(e^(2x)+1)
    return fmaf(-2.0f, __builtin_amdgcn_rcpf(1.0f + e), 1.0f);
}

// ---------------------------------------------------------------------------
// 1) Sequential exponential-smoothing scan. One wave; lanes 0..7 = series.
//    Seasonal buffer + reciprocals kept in registers via unroll-by-24
//    (t == 1 mod 24 => static indices). Loop-carried chain = 1 FMA.
// ---------------------------------------------------------------------------
__global__ __launch_bounds__(32) void es_scan_kernel(
    const float* __restrict__ x, const float* __restrict__ level_smooth,
    const float* __restrict__ seas_smooth, const float* __restrict__ init_seas,
    float* __restrict__ levels, float* __restrict__ seas)
{
    const int f = threadIdx.x;
    if (f >= F) return;

    const float alpha = sigf(level_smooth[f]);
    const float gamma = sigf(seas_smooth[f]);
    const float ooma  = 1.0f - alpha;
    const float oomg  = 1.0f - gamma;

    float buf[S], rbuf[S];
#pragma unroll
    for (int j = 0; j < S; ++j) {
        float s0 = __expf(init_seas[f * S + j]);
        buf[j]  = s0;
        rbuf[j] = __builtin_amdgcn_rcpf(s0);
    }

    const float* xf   = x      + (size_t)f * T;
    float*       levf = levels + (size_t)f * T;
    float*       seaf = seas   + (size_t)f * T;

    float level = xf[0] * rbuf[0];
    levf[0] = level;
    seaf[0] = buf[0];

    for (int tb = 1; tb < T; tb += S) {
#pragma unroll
        for (int u = 0; u < S; ++u) {
            const int t = tb + u;
            if (t >= T) break;
            const int idx = (u + 1) % S;       // tb % 24 == 1 always
            const float s  = buf[idx];
            const float rs = rbuf[idx];
            const float xt = xf[t];
            const float nl = alpha * xt * rs + ooma * level;   // critical FMA
            const float rnl = __builtin_amdgcn_rcpf(nl);        // 24-step slack
            const float ns  = gamma * xt * rnl + oomg * s;
            buf[idx]  = ns;
            rbuf[idx] = __builtin_amdgcn_rcpf(ns);
            levf[t] = nl;
            seaf[t] = s;                        // reference stores PRE-update s
            level = nl;
        }
    }
}

// ---------------------------------------------------------------------------
// 2) Parallel logs: ll = log(levels), lsd = log(seas), lx = log(x)
// ---------------------------------------------------------------------------
__global__ __launch_bounds__(256) void logs_kernel(
    const float* __restrict__ levels, const float* __restrict__ seas,
    const float* __restrict__ x,
    float* __restrict__ ll, float* __restrict__ lsd, float* __restrict__ lx)
{
    int i = blockIdx.x * blockDim.x + threadIdx.x;
    if (i < F * T) {
        ll[i]  = __logf(levels[i]);
        lsd[i] = __logf(seas[i]);
        lx[i]  = __logf(x[i]);
    }
}

// ---------------------------------------------------------------------------
// 3) lvp_losses[f] = mean_t (ll[t+2]-2ll[t+1]+ll[t])^2 * LVP
// ---------------------------------------------------------------------------
__global__ __launch_bounds__(256) void lvp_kernel(
    const float* __restrict__ ll, float* __restrict__ dout_lvp)
{
    __shared__ float red[256];
    const int f   = blockIdx.x;
    const int tid = threadIdx.x;
    const float* llf = ll + (size_t)f * T;
    float s = 0.0f;
    for (int t = tid; t < T - 2; t += 256) {
        float dd = llf[t + 2] - 2.0f * llf[t + 1] + llf[t];
        s += dd * dd;
    }
    red[tid] = s;
    __syncthreads();
    for (int off = 128; off > 0; off >>= 1) {
        if (tid < off) red[tid] += red[tid + off];
        __syncthreads();
    }
    if (tid == 0) dout_lvp[f] = red[0] * (1.0f / (float)(T - 2));  // LVP = 1.0
}

// ---------------------------------------------------------------------------
// 4) labels[n][o] = lx[0][n+W+o] - ll[0][W+n] - lsd[0][n+W+o]
// ---------------------------------------------------------------------------
__global__ __launch_bounds__(256) void labels_kernel(
    const float* __restrict__ ll, const float* __restrict__ lsd,
    const float* __restrict__ lx, float* __restrict__ dout_labels)
{
    int i = blockIdx.x * blockDim.x + threadIdx.x;
    if (i < N * On) {
        int n = i / On, o = i % On;
        dout_labels[i] = lx[n + Wn + o] - ll[Wn + n] - lsd[n + Wn + o];
    }
}

// ---------------------------------------------------------------------------
// 5) seq[w][n][f] (fp16, NPAD rows) = lx - ll(level@W+n) - lsd + noise
//    Thread map: f fastest (tid&7) so each wave writes 64 contiguous bytes.
//    Reads reuse each noise cacheline over 16 consecutive w iterations.
// ---------------------------------------------------------------------------
__global__ __launch_bounds__(256) void seq_kernel(
    const float* __restrict__ noise, const float* __restrict__ ll,
    const float* __restrict__ lsd, const float* __restrict__ lx,
    _Float16* __restrict__ seq)
{
    const int f = threadIdx.x & 7;
    const int i = threadIdx.x >> 3;
    const int n = blockIdx.x * 32 + i;
    const bool valid = (n < N);

    const size_t fb = (size_t)f * T;
    const float  lw = valid ? ll[fb + Wn + n] : 0.0f;
    const float* lxp  = lx  + fb + n;
    const float* lsdp = lsd + fb + n;
    const float* np   = noise + (size_t)f * N * Wn + (size_t)(valid ? n : 0) * Wn;

    for (int w = 0; w < Wn; ++w) {
        float v = 0.0f;
        if (valid) v = lxp[w] - lw - lsdp[w] + np[w];
        seq[((size_t)w * NPAD + n) * 8 + f] = (_Float16)v;
    }
}

// ---------------------------------------------------------------------------
// 6) Weight prep: transpose+pack to f16 "BT" layouts: BT[col][k].
//    WcT: 256 x 96  ([Wh(64) ; Wx(8) ; 0(24)] per column)
//    WtT: 64 x 64,  WlT: 48 x 64
// ---------------------------------------------------------------------------
__global__ __launch_bounds__(256) void prep_weights_kernel(
    const float* __restrict__ Wx, const float* __restrict__ Wh,
    const float* __restrict__ Wt, const float* __restrict__ Wl,
    _Float16* __restrict__ WcT, _Float16* __restrict__ WtT,
    _Float16* __restrict__ WlT)
{
    const int total = GC * KC + H * H + On * H;
    for (int i = blockIdx.x * blockDim.x + threadIdx.x; i < total;
         i += gridDim.x * blockDim.x) {
        if (i < GC * KC) {
            int col = i / KC, k = i % KC;
            float v = (k < 64) ? Wh[k * GC + col]
                    : (k < 72) ? Wx[(k - 64) * GC + col] : 0.0f;
            WcT[i] = (_Float16)v;
        } else if (i < GC * KC + H * H) {
            int j = i - GC * KC;
            int col = j / H, k = j % H;
            WtT[j] = (_Float16)Wt[k * H + col];
        } else {
            int j = i - GC * KC - H * H;
            int col = j / H, k = j % H;
            WlT[j] = (_Float16)Wl[k * On + col];
        }
    }
}

// ---------------------------------------------------------------------------
// 7) Fused LSTM (168 steps) + head, WMMA f32<-f16 16x16x32.
//    Block: 256 threads = 8 waves; ROWS=64 batch rows.
//    Wave w owns gate columns [32w, 32w+32): B frags resident in VGPRs.
//    h round-trips LDS (f16) to convert D-layout -> A-layout each step.
//    x fragments double-buffered: next step's global loads issue before the
//    barrier so their latency hides behind the gating phase.
// ---------------------------------------------------------------------------
__global__ __launch_bounds__(256) void lstm_kernel(
    const _Float16* __restrict__ seq, const _Float16* __restrict__ WcT,
    const float* __restrict__ b,
    const _Float16* __restrict__ WtT, const float* __restrict__ bt,
    const _Float16* __restrict__ WlT, const float* __restrict__ bl,
    float* __restrict__ out)
{
    __shared__ float     g_lds[ROWS * GC];      // 64 KB gate pre-activations
    __shared__ _Float16  h_lds[ROWS * H];       // 8 KB hidden state (f16)

    const int tid  = threadIdx.x;
    const int lane = tid & 31;
    const int wv   = tid >> 5;          // wave 0..7
    const int lm   = lane & 15;         // matrix row / column within tile
    const int hi   = lane >> 4;         // K-half select (A & B layouts)
    const int n0   = blockIdx.x * ROWS;

    // --- resident B fragments for this wave's 2 gate col-tiles x 3 K-tiles ---
    AF bf[2][3];
    const int gcol0 = wv * 32;
#pragma unroll
    for (int j = 0; j < 2; ++j) {
        const _Float16* base = WcT + (size_t)(gcol0 + j * 16 + lm) * KC + hi * 8;
#pragma unroll
        for (int kf = 0; kf < 3; ++kf) {
            bf[j][kf].q[0] = *(const u32x4*)(base + kf * 32);
            bf[j][kf].q[1] = *(const u32x4*)(base + kf * 32 + 16);
        }
    }
    const float biasv[2] = { b[gcol0 + lm], b[gcol0 + 16 + lm] };

    // --- init h = 0 (LDS) and c = 0 (regs: 16 elems/thread) ---
    for (int i = tid; i < ROWS * H; i += 256) h_lds[i] = (_Float16)0.0f;
    float creg[16];
#pragma unroll
    for (int k = 0; k < 16; ++k) creg[k] = 0.0f;

    const u32x4 zz = {0u, 0u, 0u, 0u};

    // x fragment double buffer (lanes 0..15 carry data; lanes 16..31 zero)
    u32x4 xf[ROWS / 16];
#pragma unroll
    for (int rt = 0; rt < ROWS / 16; ++rt) {
        xf[rt] = zz;
        if (hi == 0)
            xf[rt] = *(const u32x4*)(seq + ((size_t)0 * NPAD +
                                            (n0 + rt * 16 + lm)) * 8);
    }
    __syncthreads();

    for (int step = 0; step < Wn; ++step) {
        // ---- matmul phase: g = [h | x | 0] @ Wc  (K = 96) ----
#pragma unroll
        for (int rt = 0; rt < ROWS / 16; ++rt) {
            AF a0, a1, a2;
            const _Float16* hrow = h_lds + (rt * 16 + lm) * H;
            a0.q[0] = *(const u32x4*)(hrow + hi * 8);          // K  0..31 of h
            a0.q[1] = *(const u32x4*)(hrow + 16 + hi * 8);
            a1.q[0] = *(const u32x4*)(hrow + 32 + hi * 8);     // K 32..63 of h
            a1.q[1] = *(const u32x4*)(hrow + 48 + hi * 8);
            a2.q[0] = xf[rt];                                   // K 64..95: x|0
            a2.q[1] = zz;

#pragma unroll
            for (int j = 0; j < 2; ++j) {
                v8f acc;
#pragma unroll
                for (int r = 0; r < 8; ++r) acc[r] = biasv[j];
                acc = __builtin_amdgcn_wmma_f32_16x16x32_f16(
                    false, a0.v, false, bf[j][0].v, (short)0, acc, false, false);
                acc = __builtin_amdgcn_wmma_f32_16x16x32_f16(
                    false, a1.v, false, bf[j][1].v, (short)0, acc, false, false);
                acc = __builtin_amdgcn_wmma_f32_16x16x32_f16(
                    false, a2.v, false, bf[j][2].v, (short)0, acc, false, false);
                const int col = gcol0 + j * 16 + lm;
#pragma unroll
                for (int r = 0; r < 8; ++r)
                    g_lds[(rt * 16 + r + hi * 8) * GC + col] = acc[r];
            }
        }

        // prefetch next step's x fragments (consumed after next barrier)
        if (step + 1 < Wn) {
#pragma unroll
            for (int rt = 0; rt < ROWS / 16; ++rt) {
                if (hi == 0)
                    xf[rt] = *(const u32x4*)(seq + ((size_t)(step + 1) * NPAD +
                                                    (n0 + rt * 16 + lm)) * 8);
            }
        }
        __syncthreads();

        // ---- gating phase: all 256 threads, 16 (row,col) elems each ----
#pragma unroll
        for (int k = 0; k < 16; ++k) {
            const int e   = tid + 256 * k;
            const int row = e >> 6;
            const int col = e & 63;
            const float gi = g_lds[row * GC + col];
            const float gf = g_lds[row * GC + col + 64];
            const float gg = g_lds[row * GC + col + 128];
            const float go = g_lds[row * GC + col + 192];
            const float cc = sigf(gf) * creg[k] + sigf(gi) * tanh_fast(gg);
            creg[k] = cc;
            h_lds[row * H + col] = (_Float16)(sigf(go) * tanh_fast(cc));
        }
        __syncthreads();
    }

    // ---- head phase 1: t1 = tanh(h @ Wt + bt), f16 into reused g_lds ----
    _Float16* t1 = (_Float16*)g_lds;
    for (int job = wv; job < 16; job += 8) {
        const int rt = job >> 2, jt = job & 3;
        AF a0, a1, b0, b1;
        const _Float16* hrow = h_lds + (rt * 16 + lm) * H;
        a0.q[0] = *(const u32x4*)(hrow + hi * 8);
        a0.q[1] = *(const u32x4*)(hrow + 16 + hi * 8);
        a1.q[0] = *(const u32x4*)(hrow + 32 + hi * 8);
        a1.q[1] = *(const u32x4*)(hrow + 48 + hi * 8);
        const int col = jt * 16 + lm;
        const _Float16* wb = WtT + (size_t)col * H + hi * 8;
        b0.q[0] = *(const u32x4*)(wb);      b0.q[1] = *(const u32x4*)(wb + 16);
        b1.q[0] = *(const u32x4*)(wb + 32); b1.q[1] = *(const u32x4*)(wb + 48);
        v8f acc;
        const float bb = bt[col];
#pragma unroll
        for (int r = 0; r < 8; ++r) acc[r] = bb;
        acc = __builtin_amdgcn_wmma_f32_16x16x32_f16(
            false, a0.v, false, b0.v, (short)0, acc, false, false);
        acc = __builtin_amdgcn_wmma_f32_16x16x32_f16(
            false, a1.v, false, b1.v, (short)0, acc, false, false);
#pragma unroll
        for (int r = 0; r < 8; ++r)
            t1[(rt * 16 + r + hi * 8) * H + col] = (_Float16)tanh_fast(acc[r]);
    }
    __syncthreads();

    // ---- head phase 2: out = t1 @ Wl + bl ----
    for (int job = wv; job < 12; job += 8) {
        const int rt = job / 3, jl = job % 3;
        AF a0, a1, b0, b1;
        const _Float16* trow = t1 + (rt * 16 + lm) * H;
        a0.q[0] = *(const u32x4*)(trow + hi * 8);
        a0.q[1] = *(const u32x4*)(trow + 16 + hi * 8);
        a1.q[0] = *(const u32x4*)(trow + 32 + hi * 8);
        a1.q[1] = *(const u32x4*)(trow + 48 + hi * 8);
        const int col = jl * 16 + lm;
        const _Float16* wb = WlT + (size_t)col * H + hi * 8;
        b0.q[0] = *(const u32x4*)(wb);      b0.q[1] = *(const u32x4*)(wb + 16);
        b1.q[0] = *(const u32x4*)(wb + 32); b1.q[1] = *(const u32x4*)(wb + 48);
        v8f acc;
        const float bb = bl[col];
#pragma unroll
        for (int r = 0; r < 8; ++r) acc[r] = bb;
        acc = __builtin_amdgcn_wmma_f32_16x16x32_f16(
            false, a0.v, false, b0.v, (short)0, acc, false, false);
        acc = __builtin_amdgcn_wmma_f32_16x16x32_f16(
            false, a1.v, false, b1.v, (short)0, acc, false, false);
#pragma unroll
        for (int r = 0; r < 8; ++r) {
            const int row = n0 + rt * 16 + r + hi * 8;
            if (row < N) out[(size_t)row * On + col] = acc[r];
        }
    }
}

// ---------------------------------------------------------------------------
// Host launcher
// ---------------------------------------------------------------------------
extern "C" void kernel_launch(void* const* d_in, const int* in_sizes, int n_in,
                              void* d_out, int out_size, void* d_ws, size_t ws_size,
                              hipStream_t stream) {
    const float* x            = (const float*)d_in[0];
    const float* noise        = (const float*)d_in[1];
    const float* level_smooth = (const float*)d_in[2];
    const float* seas_smooth  = (const float*)d_in[3];
    const float* init_seas    = (const float*)d_in[4];
    const float* Wx           = (const float*)d_in[5];
    const float* Wh           = (const float*)d_in[6];
    const float* b            = (const float*)d_in[7];
    const float* Wt           = (const float*)d_in[8];
    const float* bt           = (const float*)d_in[9];
    const float* Wl           = (const float*)d_in[10];
    const float* bl           = (const float*)d_in[11];
    float* dout = (float*)d_out;

    // workspace layout (~46.2 MB total, 256-B aligned slices)
    size_t off = 0;
    auto carve = [&](size_t bytes) {
        size_t o = off;
        off += (bytes + 255) & ~(size_t)255;
        return o;
    };
    char* ws = (char*)d_ws;
    float*     levels = (float*)(ws + carve((size_t)F * T * 4));
    float*     seas   = (float*)(ws + carve((size_t)F * T * 4));
    float*     ll     = (float*)(ws + carve((size_t)F * T * 4));
    float*     lsd    = (float*)(ws + carve((size_t)F * T * 4));
    float*     lx     = (float*)(ws + carve((size_t)F * T * 4));
    _Float16*  WcT    = (_Float16*)(ws + carve((size_t)GC * KC * 2));
    _Float16*  WtT    = (_Float16*)(ws + carve((size_t)H * H * 2));
    _Float16*  WlT    = (_Float16*)(ws + carve((size_t)On * H * 2));
    _Float16*  seq    = (_Float16*)(ws + carve((size_t)Wn * NPAD * 8 * 2));
    (void)ws_size; (void)in_sizes; (void)n_in; (void)out_size;

    prep_weights_kernel<<<124, 256, 0, stream>>>(Wx, Wh, Wt, Wl, WcT, WtT, WlT);
    es_scan_kernel<<<1, 32, 0, stream>>>(x, level_smooth, seas_smooth,
                                         init_seas, levels, seas);
    logs_kernel<<<(F * T + 255) / 256, 256, 0, stream>>>(levels, seas, x,
                                                         ll, lsd, lx);
    lvp_kernel<<<F, 256, 0, stream>>>(ll, dout + (size_t)2 * N * On);
    labels_kernel<<<(N * On + 255) / 256, 256, 0, stream>>>(
        ll, lsd, lx, dout + (size_t)N * On);
    seq_kernel<<<NPAD / 32, 256, 0, stream>>>(noise, ll, lsd, lx, seq);
    lstm_kernel<<<NPAD / ROWS, 256, 0, stream>>>(seq, WcT, b, WtT, bt,
                                                 WlT, bl, dout);
}
